// BGATAttention_72267119722743
// MI455X (gfx1250) — matrix operand order repaired and checked by reading, not compiled, not run on hardware
//
#include <hip/hip_runtime.h>
#include <hip/hip_bf16.h>

// ---- problem constants -------------------------------------------------
#define B_   8
#define NU_  256
#define NA_  256
#define UD_  128
#define ED_  64
#define HID_ 512
#define H_   8
#define HD_  64
#define SLOPE_ 0.2f
#define SCALE_ 0.125f   // 1/sqrt(64)

typedef __bf16 bf16_t;
typedef __attribute__((ext_vector_type(16))) __bf16 v16bf;
typedef __attribute__((ext_vector_type(8)))  float  v8f;
typedef __attribute__((ext_vector_type(4)))  float  f32x4;   // ext-vector (builtin-friendly)
typedef __attribute__((ext_vector_type(4)))  unsigned int u32x4;
typedef __attribute__((ext_vector_type(8)))  int i32x8;
typedef __attribute__((ext_vector_type(4)))  int i32x4;

#if defined(__AMDGCN__) && __has_builtin(__builtin_amdgcn_tensor_load_to_lds)
#define USE_TDM 1
#else
#define USE_TDM 0
#endif

// ---- WMMA helpers ------------------------------------------------------
__device__ __forceinline__ v8f wmma_bf16(v16bf a, v16bf b, v8f c) {
  return __builtin_amdgcn_wmma_f32_16x16x32_bf16(
      false, a, false, b, (short)0, c, false, false);
}

__device__ __forceinline__ v16bf load_frag2(const bf16_t* p0, const bf16_t* p1) {
  union { uint4 q[2]; v16bf v; } u;
  u.q[0] = *reinterpret_cast<const uint4*>(p0);
  u.q[1] = *reinterpret_cast<const uint4*>(p1);
  return u.v;
}
// A operand (16x32, 16-bit): lane row = lane%16; K chunks {hi..hi+7, 16+hi..16+hi+7}
__device__ __forceinline__ v16bf load_a_frag(const bf16_t* rowK, int lane) {
  int hi = (lane & 16) ? 8 : 0;
  return load_frag2(rowK + hi, rowK + 16 + hi);
}
// B operand (32x16, 16-bit): lane col = lane%16; K chunks {hi..hi+7, 16+hi..16+hi+7}
__device__ __forceinline__ v16bf load_b_frag(const bf16_t* colK, int lane) {
  int kb = (lane & 16) ? 8 : 0;
  return load_frag2(colK + kb, colK + 16 + kb);
}

// ---- kernel 0a: weight transpose + f32->bf16 ---------------------------
// out[b][n][e] = (bf16) in[b][e][n]
__global__ void k_wt(const float* __restrict__ in, bf16_t* __restrict__ out,
                     int batches, int E, int N) {
  int total = batches * E * N;
  for (int i = blockIdx.x * blockDim.x + threadIdx.x; i < total;
       i += gridDim.x * blockDim.x) {
    int b = i / (E * N);
    int r = i - b * (E * N);
    int n = r / E;
    int e = r - n * E;
    out[i] = (bf16_t)in[(size_t)b * E * N + (size_t)e * N + n];
  }
}

// ---- kernel 0b: edge tensor f32 -> bf16 (read-once, non-temporal) ------
__global__ void k_cvt_bf16(const float* __restrict__ in, bf16_t* __restrict__ out,
                           long long n8) {
  for (long long i = (long long)blockIdx.x * blockDim.x + threadIdx.x; i < n8;
       i += (long long)gridDim.x * blockDim.x) {
    const f32x4* p = reinterpret_cast<const f32x4*>(in + i * 8);
    f32x4 v0 = __builtin_nontemporal_load(p);
    f32x4 v1 = __builtin_nontemporal_load(p + 1);
    union { bf16_t h[8]; uint4 q; } u;
    u.h[0] = (bf16_t)v0.x; u.h[1] = (bf16_t)v0.y;
    u.h[2] = (bf16_t)v0.z; u.h[3] = (bf16_t)v0.w;
    u.h[4] = (bf16_t)v1.x; u.h[5] = (bf16_t)v1.y;
    u.h[6] = (bf16_t)v1.z; u.h[7] = (bf16_t)v1.w;
    *reinterpret_cast<uint4*>(out + i * 8) = u.q;
  }
}

__global__ void k_zero(float* __restrict__ p, int n) {
  int i = blockIdx.x * blockDim.x + threadIdx.x;
  if (i < n) p[i] = 0.0f;
}

// ---- kernel 1: generic projection GEMM ---------------------------------
// Y[M][512] = bf16(X[M][128]) @ Wt, Wt stored [512][128] bf16 (K contiguous)
__global__ void k_proj(const float* __restrict__ X, const bf16_t* __restrict__ Wt,
                       float* __restrict__ Y) {
  __shared__ bf16_t sX[16 * 136];
  const int tid = threadIdx.x, lane = tid & 31, w = tid >> 5;
  const int m0 = blockIdx.x * 16;

  for (int i = tid; i < 16 * 128; i += 256) {
    int r = i >> 7, c = i & 127;
    sX[r * 136 + c] = (bf16_t)X[(size_t)(m0 + r) * 128 + c];
  }
  __syncthreads();

  const int colL = lane & 15;
  const int rb   = (lane & 16) ? 8 : 0;
  v8f acc[4];
  for (int nt = 0; nt < 4; ++nt)
    for (int j = 0; j < 8; ++j) acc[nt][j] = 0.0f;

  for (int s = 0; s < 4; ++s) {
    v16bf af = load_a_frag(&sX[colL * 136 + 32 * s], lane);
    for (int nt = 0; nt < 4; ++nt) {
      const bf16_t* wb = Wt + ((size_t)(w * 64 + nt * 16 + colL)) * 128 + 32 * s;
      acc[nt] = wmma_bf16(af, load_b_frag(wb, lane), acc[nt]);
    }
  }
  for (int nt = 0; nt < 4; ++nt)
    for (int r = 0; r < 8; ++r)
      Y[(size_t)(m0 + r + rb) * 512 + (w * 64 + nt * 16 + colL)] = acc[nt][r];
}

// ---- edge-tile staging into LDS ----------------------------------------
#define ESTR 72   // 64 data + 8 pad bf16 -> 144B rows (16B aligned)

#if USE_TDM
// One TDM descriptor: 3D tile (x=e:64, y=a:16 stride 64, z=u:16 stride NA*64),
// LDS padding: +4 dwords after every 32 dwords -> 144B row stride == ESTR.
__device__ __forceinline__ void tdm_load_edge_tile(const bf16_t* gbase,
                                                   unsigned lds_off) {
  unsigned long long ga = (unsigned long long)(uintptr_t)gbase;
  u32x4 g0 = {};
  g0[0] = 1u;                                            // count=1, user mode
  g0[1] = lds_off;                                       // lds_addr (bytes)
  g0[2] = (unsigned)(ga & 0xFFFFFFFFull);                // global_addr lo
  g0[3] = (unsigned)((ga >> 32) & 0x1FFFFFFull) | (2u << 30);  // hi | type=2
  i32x8 g1 = {};
  g1[0] = (1 << 16)            // data_size = 2 bytes
        | (1 << 20)            // pad_enable
        | (4 << 22)            // pad_interval code 4 -> every 32 dwords (128B)
        | (3 << 25);           // pad_amount  code 3 -> 4 dwords (16B)
  g1[1] = (64 << 16);          // tensor_dim0 = 64 (bits 63:48)
  g1[2] = (256 << 16);         // tensor_dim1 = 256 (bits 111:80 low half)
  g1[3] = (64 << 16);          // tile_dim0 = 64 (bits 127:112)
  g1[4] = 16 | (16 << 16);     // tile_dim1 = 16, tile_dim2 = 16
  g1[5] = 64;                  // tensor_dim0_stride = 64 (a step)
  g1[6] = ((NA_ * 64) & 0xFFFF) << 16;   // tensor_dim1_stride lo (u step)
  g1[7] = (NA_ * 64) >> 16;              // tensor_dim1_stride hi
  i32x4 g2 = {};
  g2[0] = 256;                 // tensor_dim2 (u extent)
  i32x4 g3 = {};
#if __clang_major__ >= 23
  i32x8 g4 = {};
  __builtin_amdgcn_tensor_load_to_lds(g0, g1, g2, g3, g4, 0);
#else
  __builtin_amdgcn_tensor_load_to_lds(g0, g1, g2, g3, 0);
#endif
}
#endif

// Fallback / generic: cooperative bf16 tile copy (row p = tid)
__device__ __forceinline__ void copy_edge_tile(const bf16_t* __restrict__ tileBase,
                                               bf16_t* __restrict__ sE, int tid) {
  const uint4* src = reinterpret_cast<const uint4*>(
      tileBase + ((size_t)(tid >> 4) * NA_ + (tid & 15)) * ED_);
  uint4* dst = reinterpret_cast<uint4*>(&sE[tid * ESTR]);
#pragma unroll
  for (int i = 0; i < 8; ++i) dst[i] = src[i];
}

__device__ __forceinline__ void stage_tile(const bf16_t* __restrict__ tileBase,
                                           bf16_t* __restrict__ sE,
                                           int tid, int w) {
#if USE_TDM
  if (w == 0) {
    tdm_load_edge_tile(tileBase, (unsigned)(uintptr_t)sE);
    __builtin_amdgcn_s_wait_tensorcnt(0);
  }
  (void)tid;
#else
  copy_edge_tile(tileBase, sE, tid);
  (void)w;
#endif
  __syncthreads();
}

// ---- kernel 2: attention scores ----------------------------------------
// grid (a_tile=16, u_tile=16, b=8); 8 waves = 8 heads share one edge tile.
__global__ void k_score(const bf16_t* __restrict__ edgeBf,
                        const float* __restrict__ Uall,
                        const float* __restrict__ Aall,
                        const bf16_t* __restrict__ WeT,
                        const float* __restrict__ av,
                        float* __restrict__ scores) {
  __shared__ bf16_t sE[256 * ESTR];         // 36 KB
  const int a0 = blockIdx.x * 16, u0 = blockIdx.y * 16, b = blockIdx.z;
  const int tid = threadIdx.x, lane = tid & 31, w = tid >> 5;   // w = head
  const int colL = lane & 15, rb = (lane & 16) ? 8 : 0;

  const bf16_t* tileBase = edgeBf + (((size_t)b * NU_ + u0) * NA_ + a0) * ED_;
  stage_tile(tileBase, sE, tid, w);

  v16bf bfr[2][4];
  for (int s = 0; s < 2; ++s)
    for (int nt = 0; nt < 4; ++nt)
      bfr[s][nt] = load_b_frag(
          WeT + ((size_t)w * 64 + nt * 16 + colL) * 64 + 32 * s, lane);
  float avr[4];
  for (int nt = 0; nt < 4; ++nt) avr[nt] = av[w * 64 + nt * 16 + colL];

  for (int t = 0; t < 16; ++t) {            // one user per M-tile, rows=antennas
    v8f acc[4];
    for (int nt = 0; nt < 4; ++nt)
      for (int j = 0; j < 8; ++j) acc[nt][j] = 0.0f;
    for (int s = 0; s < 2; ++s) {
      v16bf af = load_a_frag(&sE[(t * 16 + colL) * ESTR + 32 * s], lane);
      for (int nt = 0; nt < 4; ++nt) acc[nt] = wmma_bf16(af, bfr[s][nt], acc[nt]);
    }
    const float* Urow = Uall + ((size_t)b * NU_ + u0 + t) * 512 + w * 64;
    float rowsum[8] = {0, 0, 0, 0, 0, 0, 0, 0};
    for (int nt = 0; nt < 4; ++nt) {
      float uval = Urow[nt * 16 + colL];
      for (int r = 0; r < 8; ++r) {
        float aval = Aall[((size_t)b * NA_ + a0 + r + rb) * 512 + w * 64 +
                          nt * 16 + colL];
        float sv  = acc[nt][r] + uval + aval;
        float act = (sv >= 0.0f) ? sv : SLOPE_ * sv;
        rowsum[r] += act * avr[nt];
      }
    }
    for (int r = 0; r < 8; ++r) {
      float v = rowsum[r];
      v += __shfl_xor(v, 1, 32);
      v += __shfl_xor(v, 2, 32);
      v += __shfl_xor(v, 4, 32);
      v += __shfl_xor(v, 8, 32);
      rowsum[r] = v;
    }
    if (colL == 0) {
      float* sp = scores + (((size_t)w * B_ + b) * NU_ + (u0 + t)) * NA_ + a0 + rb;
      for (int r = 0; r < 8; ++r) sp[r] = rowsum[r] * SCALE_;
    }
  }
}

// ---- kernel 3: softmax over antennas (one wave per row of 256) ---------
__global__ void k_softmax(float* __restrict__ scores) {
  const int tid = threadIdx.x, lane = tid & 31, w = tid >> 5;
  float* p = scores + ((size_t)blockIdx.x * 8 + w) * NA_;
  float vals[8];
  float m = -1e30f;
  for (int i = 0; i < 8; ++i) { vals[i] = p[lane + i * 32]; m = fmaxf(m, vals[i]); }
  for (int s = 1; s < 32; s <<= 1) m = fmaxf(m, __shfl_xor(m, s, 32));
  float sum = 0.0f;
  for (int i = 0; i < 8; ++i) { vals[i] = __expf(vals[i] - m); sum += vals[i]; }
  for (int s = 1; s < 32; s <<= 1) sum += __shfl_xor(sum, s, 32);
  float inv = 1.0f / sum;
  for (int i = 0; i < 8; ++i) p[lane + i * 32] = vals[i] * inv;
}

// ---- kernel 4: alpha-weighted aggregation ------------------------------
__global__ void k_aggregate(const bf16_t* __restrict__ edgeBf,
                            const float* __restrict__ Uall,
                            const float* __restrict__ Aall,
                            const bf16_t* __restrict__ WeT,
                            const float* __restrict__ alpha,
                            float* __restrict__ userOut,
                            float* __restrict__ antOut) {
  __shared__ bf16_t sE[256 * ESTR];
  const int a0 = blockIdx.x * 16, u0 = blockIdx.y * 16, b = blockIdx.z;
  const int tid = threadIdx.x, lane = tid & 31, w = tid >> 5;
  const int colL = lane & 15, rb = (lane & 16) ? 8 : 0;

  const bf16_t* tileBase = edgeBf + (((size_t)b * NU_ + u0) * NA_ + a0) * ED_;
  stage_tile(tileBase, sE, tid, w);

  v16bf bfr[2][4];
  for (int s = 0; s < 2; ++s)
    for (int nt = 0; nt < 4; ++nt)
      bfr[s][nt] = load_b_frag(
          WeT + ((size_t)w * 64 + nt * 16 + colL) * 64 + 32 * s, lane);

  float ant_acc[8][4];
  for (int r = 0; r < 8; ++r)
    for (int nt = 0; nt < 4; ++nt) ant_acc[r][nt] = 0.0f;

  for (int t = 0; t < 16; ++t) {
    v8f acc[4];
    for (int nt = 0; nt < 4; ++nt)
      for (int j = 0; j < 8; ++j) acc[nt][j] = 0.0f;
    for (int s = 0; s < 2; ++s) {
      v16bf af = load_a_frag(&sE[(t * 16 + colL) * ESTR + 32 * s], lane);
      for (int nt = 0; nt < 4; ++nt) acc[nt] = wmma_bf16(af, bfr[s][nt], acc[nt]);
    }
    const float* ap = alpha + (((size_t)w * B_ + b) * NU_ + (u0 + t)) * NA_ + a0;
    float al[8];
    for (int r = 0; r < 8; ++r) al[r] = ap[r + rb];

    const float* Urow = Uall + ((size_t)b * NU_ + u0 + t) * 512 + w * 64;
    for (int nt = 0; nt < 4; ++nt) {
      float uval = Urow[nt * 16 + colL];
      float upart = 0.0f;
      for (int r = 0; r < 8; ++r) {
        float aval = Aall[((size_t)b * NA_ + a0 + r + rb) * 512 + w * 64 +
                          nt * 16 + colL];
        float ev = acc[nt][r];
        upart += al[r] * (aval + ev);          // -> user_heads
        ant_acc[r][nt] += al[r] * (uval + ev); // -> ant_heads
      }
      float v = upart + __shfl_xor(upart, 16, 32);
      if (lane < 16)
        atomicAdd(&userOut[((size_t)b * NU_ + u0 + t) * 512 + w * 64 +
                           nt * 16 + lane], v);
    }
  }
  for (int r = 0; r < 8; ++r)
    for (int nt = 0; nt < 4; ++nt)
      atomicAdd(&antOut[((size_t)b * NA_ + a0 + r + rb) * 512 + w * 64 +
                        nt * 16 + colL], ant_acc[r][nt]);
}

// ---- host launch -------------------------------------------------------
extern "C" void kernel_launch(void* const* d_in, const int* in_sizes, int n_in,
                              void* d_out, int out_size, void* d_ws, size_t ws_size,
                              hipStream_t stream) {
  const float* user_feats = (const float*)d_in[0];   // [B,NU,128]
  const float* ant_feats  = (const float*)d_in[1];   // [B,NA,128]
  const float* edge_feats = (const float*)d_in[2];   // [B,NU,NA,64]
  const float* Wu   = (const float*)d_in[3];         // [H,128,64]
  const float* Wa   = (const float*)d_in[4];         // [H,128,64]
  const float* We   = (const float*)d_in[5];         // [H,64,64]
  const float* av   = (const float*)d_in[6];         // [H,64]
  const float* Wres = (const float*)d_in[7];         // [128,512]

  float* userOut = (float*)d_out;                      // [B,NU,512]
  float* antOut  = userOut + (size_t)B_ * NU_ * HID_;  // [B,NA,512]

  char* w = (char*)d_ws;
  float* Uall   = (float*)w;  w += (size_t)B_ * NU_ * HID_ * 4;     // 4 MB
  float* Aall   = (float*)w;  w += (size_t)B_ * NA_ * HID_ * 4;     // 4 MB
  float* scores = (float*)w;  w += (size_t)H_ * B_ * NU_ * NA_ * 4; // 16 MB
  bf16_t* edgeBf = (bf16_t*)w; w += (size_t)B_ * NU_ * NA_ * ED_ * 2; // 64 MB
  bf16_t* WuT   = (bf16_t*)w; w += (size_t)H_ * HD_ * UD_ * 2;
  bf16_t* WaT   = (bf16_t*)w; w += (size_t)H_ * HD_ * UD_ * 2;
  bf16_t* WeT   = (bf16_t*)w; w += (size_t)H_ * HD_ * ED_ * 2;
  bf16_t* WresT = (bf16_t*)w; w += (size_t)HID_ * UD_ * 2;

  // weights -> transposed bf16
  k_wt<<<64, 256, 0, stream>>>(Wu,   WuT,   H_, 128, 64);
  k_wt<<<64, 256, 0, stream>>>(Wa,   WaT,   H_, 128, 64);
  k_wt<<<16, 256, 0, stream>>>(We,   WeT,   H_, 64,  64);
  k_wt<<<128, 256, 0, stream>>>(Wres, WresT, 1, 128, 512);

  // edge tensor -> bf16 once (then L2-resident for both consumer passes)
  long long n8 = (long long)B_ * NU_ * NA_ * ED_ / 8;
  k_cvt_bf16<<<8192, 256, 0, stream>>>(edge_feats, edgeBf, n8);

  // zero ant output region (user region overwritten by residual GEMM below)
  k_zero<<<(B_ * NA_ * HID_ + 255) / 256, 256, 0, stream>>>(antOut,
                                                           B_ * NA_ * HID_);
  // projections (M = 2048 rows each)
  k_proj<<<128, 256, 0, stream>>>(user_feats, WuT,   Uall);
  k_proj<<<128, 256, 0, stream>>>(ant_feats,  WaT,   Aall);
  k_proj<<<128, 256, 0, stream>>>(user_feats, WresT, userOut);  // residual

  dim3 gTile(16, 16, 8);   // (a_tile, u_tile, batch)
  k_score<<<gTile, 256, 0, stream>>>(edgeBf, Uall, Aall, WeT, av, scores);
  k_softmax<<<(H_ * B_ * NU_) / 8, 256, 0, stream>>>(scores);
  k_aggregate<<<gTile, 256, 0, stream>>>(edgeBf, Uall, Aall, WeT, scores,
                                         userOut, antOut);
}
